// Deblur_samll_60550448939592
// MI455X (gfx1250) — matrix operand order repaired and checked by reading, not compiled, not run on hardware
//
#include <hip/hip_runtime.h>
#include <hip/hip_bf16.h>
#include <math.h>

typedef __attribute__((ext_vector_type(16))) _Float16 v16h;
typedef __attribute__((ext_vector_type(8)))  _Float16 v8h;
typedef __attribute__((ext_vector_type(8)))  float    v8f;

#define DIVUP(a,b) (((a)+(b)-1)/(b))

// ---------------------------------------------------------------------------
// Pack OIHW f32 3x3 weights (C out = C in = C, C multiple of 32) into the
// CDNA5 WMMA A-fragment-friendly layout, converted to f16:
//   Wp[((kk*nIcb + icb)*C + oc)*32 + ici] = W[oc][icb*32+ici][kk/3][kk%3]
// One weight stack per blockIdx.z (for stacked resblock weight tensors).
// ---------------------------------------------------------------------------
__global__ __launch_bounds__(256)
void k_pack_w(const float* __restrict__ w, _Float16* __restrict__ wp, int C) {
  const int nIcb = C >> 5;
  const int total = 9 * C * C;
  int i = blockIdx.x * blockDim.x + threadIdx.x;
  if (i >= total) return;
  const size_t s = blockIdx.z;
  const float* wsrc = w + s * (size_t)total;
  _Float16*    wdst = wp + s * (size_t)total;
  int ici = i & 31;
  int t   = i >> 5;
  int oc  = t % C;  t /= C;
  int icb = t % nIcb;
  int kk  = t / nIcb;
  int ic  = icb * 32 + ici;
  wdst[i] = (_Float16)wsrc[((size_t)oc * C + ic) * 9 + kk];
}

// zero-fill (u32 granularity; all our f16 buffers have even element counts)
__global__ __launch_bounds__(256)
void k_fill0(unsigned int* __restrict__ p, size_t n32) {
  size_t i = (size_t)blockIdx.x * blockDim.x + threadIdx.x;
  if (i < n32) p[i] = 0u;
}
static void fill0h(_Float16* p, size_t nhalf, hipStream_t s) {
  size_t n32 = nhalf >> 1;
  k_fill0<<<dim3((unsigned)DIVUP(n32, (size_t)256)), 256, 0, s>>>(
      (unsigned int*)p, n32);
}

// ---------------------------------------------------------------------------
// WMMA implicit-GEMM 3x3 SAME conv, Cin == Cout == C (multiple of 32).
// Activations are read from an f16 NHWC companion with a 1-pixel ZERO HALO:
// layout [H+2][W+2][C], channels innermost (K-contiguous).  SAME padding
// comes from the halo data, so the K-loop is branch- and mask-free:
// per 32-K chunk it is exactly 2+2 global_load_b128 at immediate offsets
// from two fixed base addresses + one v_wmma_f32_16x16x32_f16.
//
// Epilogues:
//   F16ONLY=true : ReLU'd result as one packed-f16 b128 store (padded NHWC).
//   F16ONLY=false: f32 NCHW (+ optional residual) AND dual-write the padded
//                  f16 NHWC companion.
// Block = 128 thr = 4 waves; wave w owns a 16(oc) x 16(px) tile.
// grid.x = (W/64)*(C/16), grid.y = H, grid.z = image.
// ---------------------------------------------------------------------------
template <int C, int H, int W, bool RELU, bool RES, bool F16ONLY>
__global__ __launch_bounds__(128)
void k_wmma_conv3(const _Float16* __restrict__ Xh, const _Float16* __restrict__ Wp,
                  const float* __restrict__ bias, const float* __restrict__ res,
                  float* __restrict__ Y32, _Float16* __restrict__ Yh) {
  constexpr int HW   = H * W;
  constexpr int PW   = W + 2;
  constexpr int PHW  = (H + 2) * (W + 2);
  constexpr int nIcb = C >> 5;
  constexpr int xTiles = W >> 6;

  const int lane = threadIdx.x & 31;
  const int wv   = threadIdx.x >> 5;
  const int n    = lane & 15;   // column (pixel) / A row (oc) selector
  const int hf   = lane >> 4;   // half-wave: selects K sub-range

  const int xb = blockIdx.x % xTiles;
  const int ot = blockIdx.x / xTiles;
  const int ocBase = ot * 16;
  const int y = blockIdx.y;
  const int pxBase = xb * 64 + wv * 16;

  Xh += (size_t)blockIdx.z * PHW * C;
  Yh += (size_t)blockIdx.z * PHW * C;
  if (!F16ONLY) Y32 += (size_t)blockIdx.z * C * HW;
  if (RES)      res += (size_t)blockIdx.z * C * HW;

  // Fixed bases; all per-chunk displacements are compile-time immediates.
  const _Float16* abase = Wp + (size_t)(ocBase + n) * 32 + hf * 8;
  const _Float16* bbase = Xh + ((size_t)y * PW + (pxBase + n)) * C + hf * 8;

  v8f acc = {0.f, 0.f, 0.f, 0.f, 0.f, 0.f, 0.f, 0.f};
  union AV { v16h v; v8h h[2]; _Float16 e[16]; };

#pragma unroll
  for (int kk = 0; kk < 9; ++kk) {
    const int ky = kk / 3, kx = kk - 3 * ky;
#pragma unroll
    for (int icb = 0; icb < nIcb; ++icb) {
      AV A;
      const _Float16* ap = abase + (size_t)(kk * nIcb + icb) * C * 32;
      A.h[0] = *(const v8h*)ap;          // K local: hf*8 .. hf*8+7
      A.h[1] = *(const v8h*)(ap + 16);   // K local: hf*8+16 .. hf*8+23
      AV B;
      const _Float16* bp = bbase + (size_t)(ky * PW + kx) * C + icb * 32;
      B.h[0] = *(const v8h*)bp;          // K local: hf*8 .. hf*8+7
      B.h[1] = *(const v8h*)(bp + 16);   // K local: hf*8+16 .. hf*8+23
      acc = __builtin_amdgcn_wmma_f32_16x16x32_f16(
          false, A.v, false, B.v, (short)0, acc, false, false);
    }
  }

  // Epilogue: D VGPR r holds out[oc = ocBase + r + 8*hf][px = pxBase + n].
  // A lane's 8 results are 8 consecutive channels at one pixel.
  const int px = pxBase + n;
  const size_t sbase = (size_t)y * W + px;  // unpadded f32 NCHW coords

  float v[8];
#pragma unroll
  for (int r = 0; r < 8; ++r) {
    const int oc = ocBase + r + hf * 8;
    v[r] = acc[r] + bias[oc];
    if (RELU) v[r] = fmaxf(v[r], 0.f);
    if (RES)  v[r] += res[(size_t)oc * HW + sbase];
  }
  if (!F16ONLY) {
#pragma unroll
    for (int r = 0; r < 8; ++r)
      Y32[(size_t)(ocBase + r + hf * 8) * HW + sbase] = v[r];
  }
  union HV { v8h h; _Float16 e[8]; } hv;
#pragma unroll
  for (int r = 0; r < 8; ++r) hv.e[r] = (_Float16)v[r];
  *(v8h*)(Yh + ((size_t)(y + 1) * PW + (px + 1)) * C + ocBase + hf * 8) = hv.h;
}

// resblock stack runner: x = x + conv2(relu(conv1(x))).
// conv1: padded f16 NHWC -> padded f16 NHWC tmp (ReLU).  conv2: tmp -> f32
// NCHW state with residual, dual-writing the padded f16 NHWC companion.
template <int C, int H, int W>
static void run_res_stack(float* x32, _Float16* xh, _Float16* th,
                          const _Float16* pw1, const float* b1,
                          const _Float16* pw2, const float* b2,
                          int nblk, int N, hipStream_t stream) {
  dim3 blk(128, 1, 1);
  dim3 grd((W / 64) * (C / 16), H, N);
  const size_t ws9 = 9ULL * C * C;
  for (int i = 0; i < nblk; ++i) {
    k_wmma_conv3<C, H, W, true, false, true><<<grd, blk, 0, stream>>>(
        xh, pw1 + i * ws9, b1 + i * C, nullptr, nullptr, th);
    k_wmma_conv3<C, H, W, false, true, false><<<grd, blk, 0, stream>>>(
        th, pw2 + i * ws9, b2 + i * C, x32, x32, xh);
  }
}

// ---------------------------------------------------------------------------
// Direct conv 3 -> Cout (feature extraction). Optionally dual-writes the
// padded f16 NHWC companion (used for x0, which feeds the WMMA resblocks).
// ---------------------------------------------------------------------------
__global__ __launch_bounds__(256)
void k_conv_in3(const float* __restrict__ x, const float* __restrict__ w,
                const float* __restrict__ b, float* __restrict__ y,
                _Float16* __restrict__ yh, int Cout, int H, int W) {
  const size_t HW = (size_t)H * W;
  const size_t tot = (size_t)Cout * HW;
  size_t i = (size_t)blockIdx.x * blockDim.x + threadIdx.x;
  if (i >= tot) return;
  const int px = (int)(i % W);
  const int py = (int)((i / W) % H);
  const int oc = (int)(i / HW);
  const float* xb = x + (size_t)blockIdx.z * 3 * HW;
  float acc = b[oc];
  for (int ic = 0; ic < 3; ++ic)
    for (int ky = 0; ky < 3; ++ky) {
      const int yy = py + ky - 1;
      if (yy < 0 || yy >= H) continue;
      for (int kx = 0; kx < 3; ++kx) {
        const int xx = px + kx - 1;
        if (xx < 0 || xx >= W) continue;
        acc += xb[(size_t)ic * HW + (size_t)yy * W + xx] *
               w[((oc * 3 + ic) * 3 + ky) * 3 + kx];
      }
    }
  y[(size_t)blockIdx.z * Cout * HW + i] = acc;
  if (yh) {
    const size_t PHW = (size_t)(H + 2) * (W + 2);
    yh[((size_t)blockIdx.z * PHW + (size_t)(py + 1) * (W + 2) + (px + 1)) *
           Cout + oc] = (_Float16)acc;
  }
}

// ---------------------------------------------------------------------------
// Direct conv Cin -> 3 with sigmoid (restore heads).
// ---------------------------------------------------------------------------
__global__ __launch_bounds__(256)
void k_conv_out3_sig(const float* __restrict__ x, const float* __restrict__ w,
                     const float* __restrict__ b, float* __restrict__ y,
                     int Cin, int H, int W) {
  const size_t HW = (size_t)H * W;
  const size_t tot = 3 * HW;
  size_t i = (size_t)blockIdx.x * blockDim.x + threadIdx.x;
  if (i >= tot) return;
  const int px = (int)(i % W);
  const int py = (int)((i / W) % H);
  const int oc = (int)(i / HW);
  const float* xb = x + (size_t)blockIdx.z * Cin * HW;
  float acc = b[oc];
  for (int ic = 0; ic < Cin; ++ic) {
    const float* xc = xb + (size_t)ic * HW;
    const float* wc = w + ((size_t)oc * Cin + ic) * 9;
    for (int ky = 0; ky < 3; ++ky) {
      const int yy = py + ky - 1;
      if (yy < 0 || yy >= H) continue;
      for (int kx = 0; kx < 3; ++kx) {
        const int xx = px + kx - 1;
        if (xx < 0 || xx >= W) continue;
        acc += xc[(size_t)yy * W + xx] * wc[ky * 3 + kx];
      }
    }
  }
  y[(size_t)blockIdx.z * 3 * HW + i] = 1.f / (1.f + expf(-acc));
}

// ---------------------------------------------------------------------------
// Bilinear 1/4 resize of mid frame (t=2) of lrs[2,5,3,256,256] -> [2,3,64,64].
// jax bilinear 4x downscale (no antialias) == average of the 2x2 inner pixels.
// ---------------------------------------------------------------------------
__global__ __launch_bounds__(256)
void k_resize4(const float* __restrict__ lrs, float* __restrict__ y) {
  int i = blockIdx.x * blockDim.x + threadIdx.x;
  if (i >= 3 * 64 * 64) return;
  const int x = i & 63, yy = (i >> 6) & 63, c = i >> 12;
  const size_t b = blockIdx.z;
  const float* in = lrs + ((b * 5 + 2) * 3 + c) * (size_t)65536;
  const int r = (4 * yy + 1) * 256 + (4 * x + 1);
  const float v = 0.25f * (in[r] + in[r + 1] + in[r + 256] + in[r + 257]);
  y[(b * 3 + c) * (size_t)4096 + yy * 64 + x] = v;
}

// ---------------------------------------------------------------------------
// GAP over pix_unshuffle(x,4) of a [2,32,64,64] map -> p[2,512].
// channel c = c0*16 + r1*4 + r2 samples x[c0][4h+r1][4w+r2].
// ---------------------------------------------------------------------------
__global__ __launch_bounds__(256)
void k_gap(const float* __restrict__ x, float* __restrict__ p) {
  int i = blockIdx.x * blockDim.x + threadIdx.x;
  if (i >= 2 * 512) return;
  const int b = i >> 9, c = i & 511;
  const int c0 = c >> 4, r1 = (c >> 2) & 3, r2 = c & 3;
  const float* xb = x + ((size_t)b * 32 + c0) * 4096;
  float s = 0.f;
  for (int hh = 0; hh < 16; ++hh)
    for (int ww = 0; ww < 16; ++ww)
      s += xb[(4 * hh + r1) * 64 + 4 * ww + r2];
  p[i] = s * (1.f / 256.f);
}

// hdn[b,j] = relu((p@w1.T)[b,j] * g[j]/sqrt(1+eps) + beta[j]),  j<128
__global__ __launch_bounds__(256)
void k_mlp1(const float* __restrict__ p, const float* __restrict__ w1,
            const float* __restrict__ g, const float* __restrict__ beta,
            float* __restrict__ hdn) {
  int i = blockIdx.x * blockDim.x + threadIdx.x;
  if (i >= 2 * 128) return;
  const int b = i >> 7, j = i & 127;
  float s = 0.f;
  for (int k = 0; k < 512; ++k) s += p[b * 512 + k] * w1[j * 512 + k];
  s = s * (g[j] * rsqrtf(1.f + 1e-5f)) + beta[j];
  hdn[i] = fmaxf(s, 0.f);
}

// Wk[b,j] = (hdn@w2.T)[b,j] + b2[j],  j < 4608 (= 512*9)
__global__ __launch_bounds__(256)
void k_mlp2(const float* __restrict__ hdn, const float* __restrict__ w2,
            const float* __restrict__ b2, float* __restrict__ Wk) {
  int i = blockIdx.x * blockDim.x + threadIdx.x;
  if (i >= 2 * 4608) return;
  const int b = i / 4608, j = i - b * 4608;
  float s = b2[j];
  for (int k = 0; k < 128; ++k) s += hdn[b * 128 + k] * w2[j * 128 + k];
  Wk[i] = s;
}

// ---------------------------------------------------------------------------
// Fused: pix_unshuffle(feats,4) -> dynamic 3x3 depthwise (per-sample kernels)
// -> pix_shuffle(.,2)  ==>  fB[10,128,128,128] f32 NCHW + padded f16 NHWC.
// grid.z = bt.
// ---------------------------------------------------------------------------
__global__ __launch_bounds__(256)
void k_dw1(const float* __restrict__ feats, const float* __restrict__ Wk,
           const float* __restrict__ dbias, float* __restrict__ fB,
           _Float16* __restrict__ fBh) {
  int i = blockIdx.x * blockDim.x + threadIdx.x;
  if (i >= 128 * 128 * 128) return;
  const int x2 = i & 127, y2 = (i >> 7) & 127, c2 = i >> 14;
  const int s1 = y2 & 1, s2 = x2 & 1, h = y2 >> 1, w = x2 >> 1;
  const int cm = c2 * 4 + s1 * 2 + s2;             // mid channel (0..511)
  const int c0 = cm >> 4, r1 = (cm >> 2) & 3, r2 = cm & 3;
  const int bt = blockIdx.z, bb = bt / 5;
  const float* fb = feats + ((size_t)bt * 32 + c0) * 65536;  // [256][256]
  const float* kb = Wk + ((size_t)bb * 512 + cm) * 9;
  float acc = dbias[cm];
  for (int ky = 0; ky < 3; ++ky) {
    const int hh = h + ky - 1;
    if (hh < 0 || hh >= 64) continue;
    for (int kx = 0; kx < 3; ++kx) {
      const int ww = w + kx - 1;
      if (ww < 0 || ww >= 64) continue;
      acc += fb[(4 * hh + r1) * 256 + 4 * ww + r2] * kb[ky * 3 + kx];
    }
  }
  fB[((size_t)bt * 128 + c2) * 16384 + y2 * 128 + x2] = acc;
  fBh[((size_t)bt * 16900 + (size_t)(y2 + 1) * 130 + (x2 + 1)) * 128 + c2] =
      (_Float16)acc;                              // 130*130 = 16900 padded
}

// ---------------------------------------------------------------------------
// Fused: pix_unshuffle(fB,2) -> dynamic depthwise -> pix_shuffle(.,4)
// ==> f2[10,32,256,256] f32 NCHW + padded f16 NHWC. grid.z = bt.
// ---------------------------------------------------------------------------
__global__ __launch_bounds__(256)
void k_dw2(const float* __restrict__ fB, const float* __restrict__ Wk2,
           const float* __restrict__ dbias, float* __restrict__ f2,
           _Float16* __restrict__ f2h) {
  int i = blockIdx.x * blockDim.x + threadIdx.x;
  if (i >= 32 * 256 * 256) return;
  const int x = i & 255, y = (i >> 8) & 255, cc = i >> 16;
  const int s1 = y & 3, s2 = x & 3, h = y >> 2, w = x >> 2;
  const int cm = cc * 16 + s1 * 4 + s2;            // mid2 channel (0..511)
  const int c0 = cm >> 2, r1 = (cm >> 1) & 1, r2 = cm & 1;
  const int bt = blockIdx.z, bb = bt / 5;
  const float* fb = fB + ((size_t)bt * 128 + c0) * 16384;    // [128][128]
  const float* kb = Wk2 + ((size_t)bb * 512 + cm) * 9;
  float acc = dbias[cm];
  for (int ky = 0; ky < 3; ++ky) {
    const int hh = h + ky - 1;
    if (hh < 0 || hh >= 64) continue;
    for (int kx = 0; kx < 3; ++kx) {
      const int ww = w + kx - 1;
      if (ww < 0 || ww >= 64) continue;
      acc += fb[(2 * hh + r1) * 128 + 2 * ww + r2] * kb[ky * 3 + kx];
    }
  }
  f2[((size_t)bt * 32 + cc) * 65536 + y * 256 + x] = acc;
  f2h[((size_t)bt * 66564 + (size_t)(y + 1) * 258 + (x + 1)) * 32 + cc] =
      (_Float16)acc;                              // 258*258 = 66564 padded
}

// ---------------------------------------------------------------------------
extern "C" void kernel_launch(void* const* d_in, const int* in_sizes, int n_in,
                              void* d_out, int out_size, void* d_ws, size_t ws_size,
                              hipStream_t stream) {
  (void)in_sizes; (void)n_in; (void)out_size; (void)ws_size;
  const float* lrs    = (const float*)d_in[0];
  const float* fe_w   = (const float*)d_in[1];
  const float* fe_b   = (const float*)d_in[2];
  const float* fe1_w  = (const float*)d_in[3];
  const float* fe1_b  = (const float*)d_in[4];
  const float* A1_w1  = (const float*)d_in[5];
  const float* A1_b1  = (const float*)d_in[6];
  const float* A1_w2  = (const float*)d_in[7];
  const float* A1_b2  = (const float*)d_in[8];
  const float* A2_w1  = (const float*)d_in[9];
  const float* A2_b1  = (const float*)d_in[10];
  const float* A2_w2  = (const float*)d_in[11];
  const float* A2_b2  = (const float*)d_in[12];
  const float* B1_w1  = (const float*)d_in[13];
  const float* B1_b1  = (const float*)d_in[14];
  const float* B1_w2  = (const float*)d_in[15];
  const float* B1_b2  = (const float*)d_in[16];
  const float* B2_w1  = (const float*)d_in[17];
  const float* B2_b1  = (const float*)d_in[18];
  const float* B2_w2  = (const float*)d_in[19];
  const float* B2_b2  = (const float*)d_in[20];
  const float* cl_w   = (const float*)d_in[21];
  const float* cl_b   = (const float*)d_in[22];
  const float* cl1_w  = (const float*)d_in[23];
  const float* cl1_b  = (const float*)d_in[24];
  const float* dyd_w1 = (const float*)d_in[25];
  const float* dyd_g  = (const float*)d_in[26];
  const float* dyd_bt = (const float*)d_in[27];
  const float* dyd_w2 = (const float*)d_in[28];
  const float* dyd_b2 = (const float*)d_in[29];
  const float* dynb   = (const float*)d_in[30];

  float* out_full = (float*)d_out;                       // [2,5,3,256,256]
  float* out_r4   = out_full + 2ULL * 5 * 3 * 256 * 256; // [2,3,64,64]

  // ---- workspace carve (256B aligned) ----
  size_t off = 0;
  auto carve = [&](size_t bytes) -> void* {
    void* p = (char*)d_ws + off;
    off += (bytes + 255) & ~(size_t)255;
    return p;
  };
  const size_t BIG   = 10ULL * 128 * 128 * 128;  // f32 state elements
  const size_t PB1   = 10ULL * 130 * 130 * 128;  // padded NHWC, B1 shape
  const size_t PB2   = 10ULL * 258 * 258 * 32;   // padded NHWC, B2 shape
  const size_t PHBIG = (PB1 > PB2) ? PB1 : PB2;
  const size_t ASM   = 2ULL * 32 * 64 * 64;      // A-stage f32 elements
  const size_t ASMP  = 2ULL * 66 * 66 * 32;      // A-stage padded NHWC

  float*    bufF = (float*)carve(BIG * 4);       // feats (f32), later f2 (f32)
  float*    bufG = (float*)carve(BIG * 4);       // fB (f32 NCHW state)
  _Float16* Th   = (_Float16*)carve(PHBIG * 2);  // conv1 outputs, padded NHWC
  _Float16* Gh   = (_Float16*)carve(PHBIG * 2);  // fB padded NHWC; reused f2
  float*    bufX = (float*)carve(ASM * 4);       // x0/x00 (f32 NCHW)
  float*    bufD = (float*)carve(ASM * 4);       // delivery (f32 NCHW)
  _Float16* XhA  = (_Float16*)carve(ASMP * 2);   // x00 padded NHWC companion
  _Float16* XhD  = (_Float16*)carve(ASMP * 2);   // delivery padded NHWC
  float*    bufM = (float*)carve(2ULL * 3 * 64 * 64 * 4);  // lrs_mid4
  float*    pbuf = (float*)carve(2 * 512 * 4);
  float*    hbuf = (float*)carve(2 * 128 * 4);
  float*    WkB  = (float*)carve(2 * 4608 * 4);
  float*    Wk2B = (float*)carve(2 * 4608 * 4);
  _Float16* pA1w1 = (_Float16*)carve(10ULL * 9 * 32 * 32 * 2);
  _Float16* pA1w2 = (_Float16*)carve(10ULL * 9 * 32 * 32 * 2);
  _Float16* pA2w1 = (_Float16*)carve(10ULL * 9 * 32 * 32 * 2);
  _Float16* pA2w2 = (_Float16*)carve(10ULL * 9 * 32 * 32 * 2);
  _Float16* pB1w1 = (_Float16*)carve(3ULL * 9 * 128 * 128 * 2);
  _Float16* pB1w2 = (_Float16*)carve(3ULL * 9 * 128 * 128 * 2);
  _Float16* pB2w1 = (_Float16*)carve(1ULL * 9 * 32 * 32 * 2);
  _Float16* pB2w2 = (_Float16*)carve(1ULL * 9 * 32 * 32 * 2);

  // ---- 0) pack all resblock weights to WMMA fragment layout (f16) ----
  {
    dim3 gA(DIVUP(9 * 32 * 32, 256), 1, 10);
    k_pack_w<<<gA, 256, 0, stream>>>(A1_w1, pA1w1, 32);
    k_pack_w<<<gA, 256, 0, stream>>>(A1_w2, pA1w2, 32);
    k_pack_w<<<gA, 256, 0, stream>>>(A2_w1, pA2w1, 32);
    k_pack_w<<<gA, 256, 0, stream>>>(A2_w2, pA2w2, 32);
    dim3 gB(DIVUP(9 * 128 * 128, 256), 1, 3);
    k_pack_w<<<gB, 256, 0, stream>>>(B1_w1, pB1w1, 128);
    k_pack_w<<<gB, 256, 0, stream>>>(B1_w2, pB1w2, 128);
    dim3 gC(DIVUP(9 * 32 * 32, 256), 1, 1);
    k_pack_w<<<gC, 256, 0, stream>>>(B2_w1, pB2w1, 32);
    k_pack_w<<<gC, 256, 0, stream>>>(B2_w2, pB2w2, 32);
  }

  // ---- 1) lrs_mid4 and feature-extraction convs ----
  k_resize4<<<dim3(DIVUP(3 * 64 * 64, 256), 1, 2), 256, 0, stream>>>(lrs, bufM);
  k_conv_in3<<<dim3(DIVUP(32 * 256 * 256, 256), 1, 10), 256, 0, stream>>>(
      lrs, fe1_w, fe1_b, bufF, nullptr, 32, 256, 256);        // feats (f32 only)
  fill0h(XhA, ASMP, stream);                                  // zero halo
  k_conv_in3<<<dim3(DIVUP(32 * 64 * 64, 256), 1, 2), 256, 0, stream>>>(
      bufM, fe_w, fe_b, bufX, XhA, 32, 64, 64);               // x0 (+NHWC f16)

  // ---- 2) stage A00 (10 resblocks, 32ch @64x64) -> x00 in bufX/XhA ----
  fill0h(Th, ASMP, stream);                                   // A-layout halo
  run_res_stack<32, 64, 64>(bufX, XhA, Th, pA1w1, A1_b1, pA1w2, A1_b2, 10, 2,
                            stream);

  // ---- 3) dynamic kernels Wk from x00; depthwise -> fB (f32 + f16) ----
  k_gap<<<dim3(4), 256, 0, stream>>>(bufX, pbuf);
  k_mlp1<<<dim3(1), 256, 0, stream>>>(pbuf, dyd_w1, dyd_g, dyd_bt, hbuf);
  k_mlp2<<<dim3(DIVUP(2 * 4608, 256)), 256, 0, stream>>>(hbuf, dyd_w2, dyd_b2, WkB);
  fill0h(Gh, PB1, stream);                                    // B1-layout halo
  k_dw1<<<dim3(DIVUP(128 * 128 * 128, 256), 1, 10), 256, 0, stream>>>(
      bufF, WkB, dynb, bufG, Gh);

  // ---- 4) B1 resblocks (3 blocks, 128ch @128x128) ----
  fill0h(Th, PB1, stream);                                    // B1-layout halo
  run_res_stack<128, 128, 128>(bufG, Gh, Th, pB1w1, B1_b1, pB1w2, B1_b2, 3, 10,
                               stream);

  // ---- 5) delivery = A2 resblocks on a copy of x00 (both views) ----
  hipMemcpyAsync(bufD, bufX, ASM * 4, hipMemcpyDeviceToDevice, stream);
  hipMemcpyAsync(XhD, XhA, ASMP * 2, hipMemcpyDeviceToDevice, stream);
  fill0h(Th, ASMP, stream);                                   // back to A layout
  run_res_stack<32, 64, 64>(bufD, XhD, Th, pA2w1, A2_b1, pA2w2, A2_b2, 10, 2,
                            stream);

  // ---- 6) restore4 head ----
  k_conv_out3_sig<<<dim3(DIVUP(3 * 64 * 64, 256), 1, 2), 256, 0, stream>>>(
      bufD, cl_w, cl_b, out_r4, 32, 64, 64);

  // ---- 7) Wk2 from delivery; depthwise2 -> f2 (bufF f32 + Gh-reuse f16) ----
  k_gap<<<dim3(4), 256, 0, stream>>>(bufD, pbuf);
  k_mlp1<<<dim3(1), 256, 0, stream>>>(pbuf, dyd_w1, dyd_g, dyd_bt, hbuf);
  k_mlp2<<<dim3(DIVUP(2 * 4608, 256)), 256, 0, stream>>>(hbuf, dyd_w2, dyd_b2, Wk2B);
  fill0h(Gh, PB2, stream);                                    // B2-layout halo
  k_dw2<<<dim3(DIVUP(32 * 256 * 256, 256), 1, 10), 256, 0, stream>>>(
      bufG, Wk2B, dynb, bufF, Gh);

  // ---- 8) B2 resblock (1 block, 32ch @256x256) ----
  fill0h(Th, PB2, stream);                                    // B2-layout halo
  run_res_stack<32, 256, 256>(bufF, Gh, Th, pB2w1, B2_b1, pB2w2, B2_b2, 1, 10,
                              stream);

  // ---- 9) final head -> out ----
  k_conv_out3_sig<<<dim3(DIVUP(3 * 256 * 256, 256), 1, 10), 256, 0, stream>>>(
      bufF, cl1_w, cl1_b, out_full, 32, 256, 256);
}
